// SparseConv3d_85220741088126
// MI455X (gfx1250) — matrix operand order repaired and checked by reading, not compiled, not run on hardware
//
#include <hip/hip_runtime.h>

typedef __attribute__((ext_vector_type(16))) _Float16 v16h;
typedef __attribute__((ext_vector_type(8)))  _Float16 v8h;
typedef __attribute__((ext_vector_type(8)))  float    v8f;
typedef __attribute__((ext_vector_type(4)))  float    f4;

#define IN_C 32
#define OUT_C 64
#define DD 32
#define HH 96
#define WW 96
#define NB 2

#define SPAD 40            // padded halves per spatial row (80B -> stride-20 banks, conflict free)
#define WTILE 32           // outputs along W per workgroup
#define WHALO (WTILE + 2)  // 34 w positions incl. halo
#define NS   (3*3*WHALO)   // 306 spatial slots (3 d, 3 h, 34 w)

// ---------------------------------------------------------------------------
// Kernel 1: split f32 weights into hi/lo f16 packed directly in the WMMA
// B-fragment lane layout:  ws[tap][grp][lane][0..15]=hi halves, [16..31]=lo.
// B (32x16 f16), contiguous k order used here, same as A: lane<16 -> K = 2j+p ; lane>=16 -> K = 16+2j+p ; N = lane&15.
// ---------------------------------------------------------------------------
__global__ void pack_weights(const float* __restrict__ w, _Float16* __restrict__ wp) {
    int t = blockIdx.x * blockDim.x + threadIdx.x;   // (tap, grp, lane)
    if (t >= 27 * 4 * 32) return;
    int lane = t & 31;
    int grp  = (t >> 5) & 3;
    int tap  = t >> 7;
    int kd = tap / 9, kh = (tap / 3) % 3, kw = tap % 3;
    int oc = grp * 16 + (lane & 15);
    int kbase = (lane < 16) ? 0 : 16;
    _Float16* dst = wp + (size_t)t * 32;
    for (int j = 0; j < 8; ++j) {
        for (int p = 0; p < 2; ++p) {
            int ci = kbase + 2 * j + p;              // input channel = GEMM K
            float v = w[(((oc * IN_C + ci) * 3 + kd) * 3 + kh) * 3 + kw];
            _Float16 hi = (_Float16)v;
            _Float16 lo = (_Float16)(v - (float)hi);
            dst[j * 2 + p]      = hi;
            dst[16 + j * 2 + p] = lo;
        }
    }
}

// ---------------------------------------------------------------------------
// Kernel 2: implicit-GEMM conv3d. Workgroup = 128 threads (4 wave32).
// Tile: 32 outputs along W (two 16-wide M subtiles sharing one B fragment)
// x 64 out-channels (one 16-wide group per wave) at fixed (n, d, h).
// K loop = 27 taps x 32 channels; per tap: 6x v_wmma_f32_16x16x32_f16
// (hi*hi + hi*lo + lo*hi per subtile) -> ~f32 accuracy from f16 matrix cores.
// ---------------------------------------------------------------------------
__global__ __launch_bounds__(128)
void conv3d_wmma(const float* __restrict__ x, const _Float16* __restrict__ wp,
                 const float* __restrict__ bias, float* __restrict__ out) {
    __shared__ _Float16 xs_hi[NS * SPAD];
    __shared__ _Float16 xs_lo[NS * SPAD];

    int bid = blockIdx.x;
    int wt  = bid % (WW / WTILE);   int tmp = bid / (WW / WTILE);
    int h0  = tmp % HH;             tmp /= HH;
    int d0  = tmp % DD;             int n0 = tmp / DD;
    int wBase = wt * WTILE;

    int tid = threadIdx.x;

    // ---- stage halo tile into LDS, converting f32 -> (hi,lo) f16 once ----
    for (int e = tid; e < NS * IN_C; e += 128) {
        int c = e / NS;                  // channel-major so consecutive tids
        int s = e - c * NS;              // walk contiguous w -> coalesced
        int dd = s / (3 * WHALO);
        int r  = s - dd * (3 * WHALO);
        int hh = r / WHALO;
        int ww = r - hh * WHALO;
        int id = d0 + dd - 1, ih = h0 + hh - 1, iw = wBase + ww - 1;
        float v = 0.0f;
        if ((unsigned)id < DD && (unsigned)ih < HH && (unsigned)iw < WW)
            v = x[(((size_t)n0 * IN_C + c) * DD + id) * (HH * WW) + ih * WW + iw];
        _Float16 hi = (_Float16)v;
        _Float16 lo = (_Float16)(v - (float)hi);
        xs_hi[s * SPAD + c] = hi;
        xs_lo[s * SPAD + c] = lo;
    }
    __syncthreads();

    int lane = tid & 31;
    int grp  = tid >> 5;                 // wave id -> out-channel group
    int m    = lane & 15;                // A-matrix row = w offset in subtile
    int cb0  = (lane < 16) ? 0 : 16;     // A layout: K blocks {0..15} / {16..31}
    int cb1  = cb0 + 8;

    const _Float16* wlane = wp + ((size_t)grp * 32 + lane) * 32;

    v8f acc0 = {};                       // w offsets 0..15
    v8f acc1 = {};                       // w offsets 16..31
    #pragma unroll 3
    for (int tap = 0; tap < 27; ++tap) {
        int kd = tap / 9, kh = (tap / 3) % 3, kw = tap % 3;
        int srow0 = ((kd * 3 + kh) * WHALO + m + kw) * SPAD;
        int srow1 = srow0 + 16 * SPAD;

        union { v16h v; v8h p[2]; } ah0, al0, ah1, al1, bh, bl;
        ah0.p[0] = *(const v8h*)(&xs_hi[srow0 + cb0]);
        ah0.p[1] = *(const v8h*)(&xs_hi[srow0 + cb1]);
        al0.p[0] = *(const v8h*)(&xs_lo[srow0 + cb0]);
        al0.p[1] = *(const v8h*)(&xs_lo[srow0 + cb1]);
        ah1.p[0] = *(const v8h*)(&xs_hi[srow1 + cb0]);
        ah1.p[1] = *(const v8h*)(&xs_hi[srow1 + cb1]);
        al1.p[0] = *(const v8h*)(&xs_lo[srow1 + cb0]);
        al1.p[1] = *(const v8h*)(&xs_lo[srow1 + cb1]);

        const _Float16* wt_ptr = wlane + (size_t)tap * (4 * 32 * 32);
        bh.p[0] = *(const v8h*)(wt_ptr);
        bh.p[1] = *(const v8h*)(wt_ptr + 8);
        bl.p[0] = *(const v8h*)(wt_ptr + 16);
        bl.p[1] = *(const v8h*)(wt_ptr + 24);

        acc0 = __builtin_amdgcn_wmma_f32_16x16x32_f16(false, ah0.v, false, bh.v,
                                                      (short)0, acc0, false, false);
        acc0 = __builtin_amdgcn_wmma_f32_16x16x32_f16(false, ah0.v, false, bl.v,
                                                      (short)0, acc0, false, false);
        acc0 = __builtin_amdgcn_wmma_f32_16x16x32_f16(false, al0.v, false, bh.v,
                                                      (short)0, acc0, false, false);
        acc1 = __builtin_amdgcn_wmma_f32_16x16x32_f16(false, ah1.v, false, bh.v,
                                                      (short)0, acc1, false, false);
        acc1 = __builtin_amdgcn_wmma_f32_16x16x32_f16(false, ah1.v, false, bl.v,
                                                      (short)0, acc1, false, false);
        acc1 = __builtin_amdgcn_wmma_f32_16x16x32_f16(false, al1.v, false, bh.v,
                                                      (short)0, acc1, false, false);
    }

    // ---- epilogue: C layout gives each lane 8 contiguous w values ----
    int oc   = grp * 16 + (lane & 15);
    float bv = bias[oc];
    int wOut = wBase + ((lane < 16) ? 0 : 8);
    float* o = out + (((size_t)n0 * OUT_C + oc) * DD + d0) * (HH * WW) + h0 * WW + wOut;
    f4 a = { acc0[0] + bv, acc0[1] + bv, acc0[2] + bv, acc0[3] + bv };
    f4 b = { acc0[4] + bv, acc0[5] + bv, acc0[6] + bv, acc0[7] + bv };
    f4 c = { acc1[0] + bv, acc1[1] + bv, acc1[2] + bv, acc1[3] + bv };
    f4 d = { acc1[4] + bv, acc1[5] + bv, acc1[6] + bv, acc1[7] + bv };
    __builtin_nontemporal_store(a, (f4*)o);          // write-once stream, keep L2 for x
    __builtin_nontemporal_store(b, (f4*)(o + 4));
    __builtin_nontemporal_store(c, (f4*)(o + 16));
    __builtin_nontemporal_store(d, (f4*)(o + 20));
}

extern "C" void kernel_launch(void* const* d_in, const int* in_sizes, int n_in,
                              void* d_out, int out_size, void* d_ws, size_t ws_size,
                              hipStream_t stream) {
    const float* x    = (const float*)d_in[0];
    const float* w    = (const float*)d_in[1];
    const float* bias = (const float*)d_in[2];
    float* out        = (float*)d_out;
    _Float16* wp      = (_Float16*)d_ws;   // 27*4*32*32 halves * 2B = 216 KB

    pack_weights<<<27, 128, 0, stream>>>(w, wp);

    int nTiles = NB * DD * HH * (WW / WTILE); // 18432 workgroups
    conv3d_wmma<<<nTiles, 128, 0, stream>>>(x, wp, bias, out);
}
